// RoPEAttention_31430570672330
// MI455X (gfx1250) — compile-verified, matrix operand order checked by reading, not run on hardware
//
#include <hip/hip_runtime.h>
#include <hip/hip_bf16.h>

#define SEQ 4096
#define HID 1024
#define NH  16
#define HD  64
#define MT  4          // seq-tiles per wave in the QKV GEMM

typedef __attribute__((ext_vector_type(8)))  __bf16         v8bf;
typedef __attribute__((ext_vector_type(16))) __bf16         v16bf;
typedef __attribute__((ext_vector_type(8)))  float          v8f;
typedef __attribute__((ext_vector_type(8)))  unsigned short v8u;
typedef __attribute__((ext_vector_type(4)))  unsigned short v4u;

// ---- helpers -------------------------------------------------------------

static __device__ __forceinline__ unsigned short f2bf(float f) {
  unsigned u = __builtin_bit_cast(unsigned, f);
  u += 0x7FFFu + ((u >> 16) & 1u);          // round-to-nearest-even
  return (unsigned short)(u >> 16);
}

// ds_swizzle_b32 xor-exchange, group-of-32 mode: offset = xor<<10 | and 0x1f.
// All masks < 16 -> exchange stays within the half-wave owning a row group.
#define SWZ_F32(x, imm) \
  __builtin_bit_cast(float, __builtin_amdgcn_ds_swizzle(__builtin_bit_cast(int, (x)), (imm)))

// Load a 16x32 bf16 A-fragment (or 32x16 B-fragment, same lane pattern):
// lane L<16 : row/col = L,     K = k0+0..7  and k0+16..23
// lane L>=16: row/col = L-16,  K = k0+8..15 and k0+24..31
static __device__ __forceinline__ v16bf load_frag(const unsigned short* __restrict__ base,
                                                  int row, int ld, int k0, int lane) {
  const int klo = k0 + ((lane & 16) ? 8 : 0);
  const unsigned short* p = base + (size_t)row * ld + klo;
  v8bf lo = *reinterpret_cast<const v8bf*>(p);        // global_load_b128
  v8bf hi = *reinterpret_cast<const v8bf*>(p + 16);   // global_load_b128
  return __builtin_shufflevector(lo, hi, 0,1,2,3,4,5,6,7,8,9,10,11,12,13,14,15);
}

// Same fragment pattern but sourced from LDS (ds_load_b128 x2).
static __device__ __forceinline__ v16bf load_frag_lds(const unsigned short* p) {
  v8bf lo = *reinterpret_cast<const v8bf*>(p);
  v8bf hi = *reinterpret_cast<const v8bf*>(p + 16);
  return __builtin_shufflevector(lo, hi, 0,1,2,3,4,5,6,7,8,9,10,11,12,13,14,15);
}

static __device__ __forceinline__ v8f wmma_bf16(v16bf a, v16bf b, v8f c) {
  return __builtin_amdgcn_wmma_f32_16x16x32_bf16(false, a, false, b, (short)0, c,
                                                 false, false);
}

// Async-DMA one 32-key x 64-dim V block (4KB) from Vt into LDS.
// 8 x global_load_async_to_lds_b128: instr i, lane L copies 16B of
// dim d = i*8 + L/4, key-chunk (L%4)*8. Tracked with ASYNCcnt.
static __device__ __forceinline__ void async_copy_vblock(
    unsigned short* dst_lds, const unsigned short* __restrict__ Vt,
    int dbase, int j0, int lane) {
  const int d  = lane >> 2;         // 0..7 within each group of 8 dims
  const int ch = (lane & 3) * 8;    // key chunk (8 bf16 = 16B)
  #pragma unroll
  for (int i = 0; i < 8; ++i) {
    unsigned lds_addr =
        (unsigned)(uintptr_t)(dst_lds + ((i * 8 + d) * 32 + ch));
    unsigned long long gaddr = (unsigned long long)(uintptr_t)(
        Vt + (size_t)(dbase + i * 8 + d) * SEQ + j0 + ch);
    asm volatile("global_load_async_to_lds_b128 %0, %1, off"
                 :: "v"(lds_addr), "v"(gaddr) : "memory");
  }
}

// ---- kernel 1: f32 -> bf16 conversion ------------------------------------

__global__ __launch_bounds__(256) void cvt_bf16_kernel(const float* __restrict__ s,
                                                       unsigned short* __restrict__ d,
                                                       int n) {
  int i = (blockIdx.x * 256 + threadIdx.x) * 4;
  if (i + 3 < n) {
    float4 f = *reinterpret_cast<const float4*>(s + i);
    v4u o;
    o[0] = f2bf(f.x); o[1] = f2bf(f.y); o[2] = f2bf(f.z); o[3] = f2bf(f.w);
    *reinterpret_cast<v4u*>(d + i) = o;
  }
}

// ---- kernel 2: fused QKV projection + RoPE -------------------------------

__global__ __launch_bounds__(32) void qkv_rope_kernel(
    const unsigned short* __restrict__ xb,
    const unsigned short* __restrict__ wqb,
    const unsigned short* __restrict__ wkb,
    const unsigned short* __restrict__ wvb,
    unsigned short* __restrict__ Qr,
    unsigned short* __restrict__ Kr,
    unsigned short* __restrict__ Vt) {
  const int lane = threadIdx.x;
  const int tn = blockIdx.x;            // output-feature tile 0..63
  const int tg = blockIdx.y;            // group of MT sequence tiles 0..63
  const int nl = lane & 15;
  const int rB = tn * 16 + nl;

  v8f accq[MT], acck[MT], accv[MT];
  #pragma unroll
  for (int mi = 0; mi < MT; ++mi) {
    accq[mi] = (v8f){0,0,0,0,0,0,0,0};
    acck[mi] = (v8f){0,0,0,0,0,0,0,0};
    accv[mi] = (v8f){0,0,0,0,0,0,0,0};
  }

  for (int k0 = 0; k0 < HID; k0 += 32) {
    v16bf bq = load_frag(wqb, rB, HID, k0, lane);
    v16bf bk = load_frag(wkb, rB, HID, k0, lane);
    v16bf bv = load_frag(wvb, rB, HID, k0, lane);
    #pragma unroll
    for (int mi = 0; mi < MT; ++mi) {
      v16bf a = load_frag(xb, (tg * MT + mi) * 16 + nl, HID, k0, lane);
      accq[mi] = wmma_bf16(a, bq, accq[mi]);
      acck[mi] = wmma_bf16(a, bk, acck[mi]);
      accv[mi] = wmma_bf16(a, bv, accv[mi]);
    }
  }

  const int o   = tn * 16 + nl;
  const int hi8 = (lane >> 4) * 8;
  // inv_freq = theta^(-2j/64), j = (o%64)>>1 ; log2(10000)/64 = 0.2076205059
  const float inv_f = exp2f(-(float)((o & 63) & ~1) * 0.20762050593f);
  const bool  odd   = (o & 1) != 0;

  #pragma unroll
  for (int mi = 0; mi < MT; ++mi) {
    const int s_base = (tg * MT + mi) * 16 + hi8;
    v8u vpack;
    #pragma unroll
    for (int r = 0; r < 8; ++r) {
      float ang = (float)(s_base + r) * inv_f;
      float c  = __cosf(ang);
      float sn = __sinf(ang);
      float vq = accq[mi][r], vk = acck[mi][r];
      float pq = SWZ_F32(vq, 0x041F);   // RoPE pair partner: lane xor 1
      float pk = SWZ_F32(vk, 0x041F);
      float oq = odd ? (pq * sn + vq * c) : (vq * c - pq * sn);
      float ok = odd ? (pk * sn + vk * c) : (vk * c - pk * sn);
      Qr[(size_t)(s_base + r) * HID + o] = f2bf(oq);
      Kr[(size_t)(s_base + r) * HID + o] = f2bf(ok);
      vpack[r] = f2bf(accv[mi][r]);
    }
    *reinterpret_cast<v8u*>(Vt + (size_t)o * SEQ + s_base) = vpack;
  }
}

// ---- kernel 3: flash attention -------------------------------------------
// K fragments double-buffered in VGPRs; V blocks double-buffered in LDS via
// the async-DMA path (ASYNCcnt). Row reductions via ds_swizzle xor patterns.

__global__ __launch_bounds__(32) void attn_kernel(
    const unsigned short* __restrict__ Qr,
    const unsigned short* __restrict__ Kr,
    const unsigned short* __restrict__ Vt,
    float* __restrict__ out) {
  __shared__ unsigned short Pb[16 * 32];        // P tile: C-layout -> A-layout
  __shared__ unsigned short Vb[2][64 * 32];     // V block double buffer [d][key]

  const int lane = threadIdx.x;
  const int h    = blockIdx.x;             // head 0..15
  const int tq   = blockIdx.y;             // query tile 0..255
  const int nl   = lane & 15;
  const int hi8  = (lane >> 4) * 8;
  const int rQ   = tq * 16 + nl;

  v16bf aq0 = load_frag(Qr, rQ, HID, h * HD,      lane);
  v16bf aq1 = load_frag(Qr, rQ, HID, h * HD + 32, lane);

  v8f acc[4];
  #pragma unroll
  for (int t = 0; t < 4; ++t) acc[t] = (v8f){0,0,0,0,0,0,0,0};
  v8f m_run = (v8f){-1e30f,-1e30f,-1e30f,-1e30f,-1e30f,-1e30f,-1e30f,-1e30f};
  v8f l_run = (v8f){0,0,0,0,0,0,0,0};

  // prime pipelines: K block 0 into registers, V block 0 into LDS buf 0
  v16bf kb0 = load_frag(Kr, nl,      HID, h * HD,      lane);
  v16bf kb1 = load_frag(Kr, nl,      HID, h * HD + 32, lane);
  v16bf kb2 = load_frag(Kr, 16 + nl, HID, h * HD,      lane);
  v16bf kb3 = load_frag(Kr, 16 + nl, HID, h * HD + 32, lane);
  async_copy_vblock(&Vb[0][0], Vt, h * HD, 0, lane);

  int cur = 0;
  for (int j0 = 0; j0 < SEQ; j0 += 32) {
    // ---- issue next-block prefetches (K -> VGPRs, V -> LDS async) ----
    const int jn  = j0 + 32;
    const int jnv = (jn < SEQ) ? jn : SEQ - 32;   // clamp: stay inside Vt
    v16bf kn0 = load_frag(Kr, jn + nl,      HID, h * HD,      lane);
    v16bf kn1 = load_frag(Kr, jn + nl,      HID, h * HD + 32, lane);
    v16bf kn2 = load_frag(Kr, jn + 16 + nl, HID, h * HD,      lane);
    v16bf kn3 = load_frag(Kr, jn + 16 + nl, HID, h * HD + 32, lane);
    async_copy_vblock(&Vb[cur ^ 1][0], Vt, h * HD, jnv, lane);
    __builtin_prefetch(Kr + (size_t)(j0 + 64) * HID + h * HD, 0, 1);

    // ---- scores S = Q * K^T for 32 keys (two 16x16 tiles, K-dim 64) ----
    v8f s0 = (v8f){0,0,0,0,0,0,0,0};
    v8f s1 = (v8f){0,0,0,0,0,0,0,0};
    s0 = wmma_bf16(aq0, kb0, s0);
    s0 = wmma_bf16(aq1, kb1, s0);
    s1 = wmma_bf16(aq0, kb2, s1);
    s1 = wmma_bf16(aq1, kb3, s1);

    // ---- online softmax; row reductions via ds_swizzle xor exchanges ----
    float rm[8], rs[8], corr[8];
    #pragma unroll
    for (int e = 0; e < 8; ++e) {
      s0[e] *= 0.125f;  s1[e] *= 0.125f;          // 1/sqrt(64)
      rm[e] = fmaxf(s0[e], s1[e]);
    }
    #pragma unroll
    for (int e = 0; e < 8; ++e) {
      rm[e] = fmaxf(rm[e], SWZ_F32(rm[e], 0x041F));
      rm[e] = fmaxf(rm[e], SWZ_F32(rm[e], 0x081F));
      rm[e] = fmaxf(rm[e], SWZ_F32(rm[e], 0x101F));
      rm[e] = fmaxf(rm[e], SWZ_F32(rm[e], 0x201F));
    }
    #pragma unroll
    for (int e = 0; e < 8; ++e) {
      float mn = fmaxf(m_run[e], rm[e]);
      corr[e]  = __expf(m_run[e] - mn);
      m_run[e] = mn;
      s0[e] = __expf(s0[e] - mn);
      s1[e] = __expf(s1[e] - mn);
      rs[e] = s0[e] + s1[e];
    }
    #pragma unroll
    for (int e = 0; e < 8; ++e) {
      rs[e] += SWZ_F32(rs[e], 0x041F);
      rs[e] += SWZ_F32(rs[e], 0x081F);
      rs[e] += SWZ_F32(rs[e], 0x101F);
      rs[e] += SWZ_F32(rs[e], 0x201F);
    }
    #pragma unroll
    for (int e = 0; e < 8; ++e) {
      l_run[e] = l_run[e] * corr[e] + rs[e];
      acc[0][e] *= corr[e]; acc[1][e] *= corr[e];
      acc[2][e] *= corr[e]; acc[3][e] *= corr[e];
    }

    // ---- re-layout P (C-layout) into A-layout via LDS ----
    #pragma unroll
    for (int r = 0; r < 8; ++r) {
      Pb[(r + hi8) * 32 + nl]      = f2bf(s0[r]);
      Pb[(r + hi8) * 32 + 16 + nl] = f2bf(s1[r]);
    }
    __builtin_amdgcn_wave_barrier();
    v16bf ap = load_frag_lds(Pb + nl * 32 + ((lane & 16) ? 8 : 0));
    __builtin_amdgcn_wave_barrier();

    // ---- drain current V block's async copies (next block may stay in
    //      flight: async completions are in-order, <=8 left == current done)
    asm volatile("s_wait_asynccnt 0x8" ::: "memory");

    // ---- O += P (16x32) * V (32x64) from LDS ----
    const unsigned short* vb = &Vb[cur][0];
    const int klo = (lane & 16) ? 8 : 0;
    #pragma unroll
    for (int t = 0; t < 4; ++t) {
      v16bf bv = load_frag_lds(vb + (t * 16 + nl) * 32 + klo);
      acc[t] = wmma_bf16(ap, bv, acc[t]);
    }

    // rotate double buffers
    kb0 = kn0; kb1 = kn1; kb2 = kn2; kb3 = kn3;
    cur ^= 1;
  }

  // ---- epilogue: normalize and store f32 output ----
  float invl[8];
  #pragma unroll
  for (int e = 0; e < 8; ++e) invl[e] = 1.0f / l_run[e];
  #pragma unroll
  for (int t = 0; t < 4; ++t) {
    #pragma unroll
    for (int r = 0; r < 8; ++r) {
      out[(size_t)(tq * 16 + hi8 + r) * HID + h * HD + t * 16 + nl] =
          acc[t][r] * invl[r];
    }
  }
}

// ---- host launch ---------------------------------------------------------

extern "C" void kernel_launch(void* const* d_in, const int* in_sizes, int n_in,
                              void* d_out, int out_size, void* d_ws, size_t ws_size,
                              hipStream_t stream) {
  const float* x  = (const float*)d_in[0];
  const float* wq = (const float*)d_in[1];
  const float* wk = (const float*)d_in[2];
  const float* wv = (const float*)d_in[3];
  float* out = (float*)d_out;

  unsigned short* ws  = (unsigned short*)d_ws;
  unsigned short* xb  = ws;                       // 4096*1024
  unsigned short* wqb = xb  + (size_t)SEQ * HID;  // 1024*1024
  unsigned short* wkb = wqb + (size_t)HID * HID;
  unsigned short* wvb = wkb + (size_t)HID * HID;
  unsigned short* Qr  = wvb + (size_t)HID * HID;  // 4096*1024 (RoPE'd, bf16)
  unsigned short* Kr  = Qr  + (size_t)SEQ * HID;
  unsigned short* Vt  = Kr  + (size_t)SEQ * HID;  // transposed V (feature-major)

  const int nx = SEQ * HID, nw = HID * HID;
  cvt_bf16_kernel<<<(nx / 4 + 255) / 256, 256, 0, stream>>>(x,  xb,  nx);
  cvt_bf16_kernel<<<(nw / 4 + 255) / 256, 256, 0, stream>>>(wq, wqb, nw);
  cvt_bf16_kernel<<<(nw / 4 + 255) / 256, 256, 0, stream>>>(wk, wkb, nw);
  cvt_bf16_kernel<<<(nw / 4 + 255) / 256, 256, 0, stream>>>(wv, wvb, nw);

  qkv_rope_kernel<<<dim3(HID / 16, SEQ / 16 / MT), 32, 0, stream>>>(
      xb, wqb, wkb, wvb, Qr, Kr, Vt);

  attn_kernel<<<dim3(NH, SEQ / 16), 32, 0, stream>>>(Qr, Kr, Vt, out);
}
